// AqtDotGeneral_75892072120646
// MI455X (gfx1250) — 8-bit matrix operand order repaired and checked by reading, compiled, not run on hardware
//
#include <hip/hip_runtime.h>
#include <hip/hip_bf16.h>
#include <stdint.h>

typedef __attribute__((ext_vector_type(8))) int v8i;

#define K_DIM 1024
#define N_DIM 1024
#define M_DIM 32768

__global__ __launch_bounds__(256) void quant_lhs_kernel(
    const float* __restrict__ lhs,
    int8_t* __restrict__ qA,
    float* __restrict__ lhs_scale) {
  __shared__ float red[256];
  const int r = blockIdx.x;
  const int t = threadIdx.x;

  const float4 v = *(const float4*)(lhs + (size_t)r * K_DIM + (size_t)t * 4);
  float m = fmaxf(fmaxf(fabsf(v.x), fabsf(v.y)), fmaxf(fabsf(v.z), fabsf(v.w)));
  red[t] = m;
  __syncthreads();
#pragma unroll
  for (int s = 128; s > 0; s >>= 1) {
    if (t < s) red[t] = fmaxf(red[t], red[t + s]);
    __syncthreads();
  }
  float amax = red[0];
  if (amax == 0.0f) amax = 1.0f;
  if (t == 0) lhs_scale[r] = amax * (1.0f / 127.0f);
  const float inv = 127.0f / amax;

  int q0 = (int)rintf(v.x * inv);
  int q1 = (int)rintf(v.y * inv);
  int q2 = (int)rintf(v.z * inv);
  int q3 = (int)rintf(v.w * inv);
  q0 = max(-127, min(127, q0));
  q1 = max(-127, min(127, q1));
  q2 = max(-127, min(127, q2));
  q3 = max(-127, min(127, q3));
  uint32_t p = (uint32_t)(q0 & 0xff) | ((uint32_t)(q1 & 0xff) << 8) |
               ((uint32_t)(q2 & 0xff) << 16) | ((uint32_t)(q3 & 0xff) << 24);
  *(uint32_t*)(qA + (size_t)r * K_DIM + (size_t)t * 4) = p;
}

__global__ __launch_bounds__(256) void quant_rhs_kernel(
    const float* __restrict__ rhs,
    int8_t* __restrict__ qBt,
    float* __restrict__ rhs_scale) {
  const int col = blockIdx.x * blockDim.x + threadIdx.x;
  float amax = 0.0f;
  for (int k = 0; k < K_DIM; ++k)
    amax = fmaxf(amax, fabsf(rhs[(size_t)k * N_DIM + col]));
  if (amax == 0.0f) amax = 1.0f;
  rhs_scale[col] = amax * (1.0f / 127.0f);
  const float inv = 127.0f / amax;
  for (int k = 0; k < K_DIM; k += 4) {
    uint32_t p = 0;
#pragma unroll
    for (int j = 0; j < 4; ++j) {
      int q = (int)rintf(rhs[(size_t)(k + j) * N_DIM + col] * inv);
      q = max(-127, min(127, q));
      p |= ((uint32_t)(q & 0xff)) << (8 * j);
    }
    *(uint32_t*)(qBt + (size_t)col * K_DIM + k) = p;
  }
}

__global__ __launch_bounds__(256) void int8_gemm_dequant_kernel(
    const int8_t* __restrict__ qA,
    const int8_t* __restrict__ qBt,
    const float* __restrict__ lhs_scale,
    const float* __restrict__ rhs_scale,
    float* __restrict__ out) {
  const int lane = threadIdx.x & 31;
  const int wave = threadIdx.x >> 5;
  const int grp  = lane >> 4;
  const int l16  = lane & 15;

  const int waveM = blockIdx.y * 128 + (wave & 3) * 32;
  const int waveN = blockIdx.x * 128 + (wave >> 2) * 64;

  v8i acc[2][4] = {};

  for (int kk = 0; kk < K_DIM; kk += 64) {
    v8i a[2];
    v8i b[4];
#pragma unroll
    for (int im = 0; im < 2; ++im) {
      const int m = waveM + im * 16 + l16;
      const int8_t* base = qA + (size_t)m * K_DIM + kk + grp * 16;
#pragma unroll
      for (int j = 0; j < 4; ++j) {
        const int2 d = *(const int2*)(base + (j & 1) * 8 + (j >> 1) * 32);
        a[im][2 * j]     = d.x;
        a[im][2 * j + 1] = d.y;
      }
    }
#pragma unroll
    for (int in = 0; in < 4; ++in) {
      const int n = waveN + in * 16 + l16;
      const int8_t* base = qBt + (size_t)n * K_DIM + kk + grp * 16;
      const int4 lo = *(const int4*)(base);
      const int4 hi = *(const int4*)(base + 32);
      b[in][0] = lo.x; b[in][1] = lo.y; b[in][2] = lo.z; b[in][3] = lo.w;
      b[in][4] = hi.x; b[in][5] = hi.y; b[in][6] = hi.z; b[in][7] = hi.w;
    }
#pragma unroll
    for (int im = 0; im < 2; ++im)
#pragma unroll
      for (int in = 0; in < 4; ++in)
        acc[im][in] = __builtin_amdgcn_wmma_i32_16x16x64_iu8(
             true, a[im],  true, b[in], acc[im][in],
             false,  false);
  }

#pragma unroll
  for (int im = 0; im < 2; ++im) {
#pragma unroll
    for (int v = 0; v < 8; ++v) {
      const int m = waveM + im * 16 + v + 8 * grp;
      const float ls = lhs_scale[m];
#pragma unroll
      for (int in = 0; in < 4; ++in) {
        const int n = waveN + in * 16 + l16;
        out[(size_t)m * N_DIM + n] = (float)acc[im][in][v] * ls * rhs_scale[n];
      }
    }
  }
}

extern "C" void kernel_launch(void* const* d_in, const int* in_sizes, int n_in,
                              void* d_out, int out_size, void* d_ws, size_t ws_size,
                              hipStream_t stream) {
  const float* lhs = (const float*)d_in[0];
  const float* rhs = (const float*)d_in[1];
  float* out = (float*)d_out;

  int8_t* qA  = (int8_t*)d_ws;
  int8_t* qBt = qA + (size_t)M_DIM * K_DIM;
  float* lhs_scale = (float*)(qBt + (size_t)N_DIM * K_DIM);
  float* rhs_scale = lhs_scale + M_DIM;

  quant_lhs_kernel<<<M_DIM, 256, 0, stream>>>(lhs, qA, lhs_scale);
  quant_rhs_kernel<<<N_DIM / 256, 256, 0, stream>>>(rhs, qBt, rhs_scale);

  dim3 grid(N_DIM / 128, M_DIM / 128);
  int8_gemm_dequant_kernel<<<grid, 256, 0, stream>>>(qA, qBt, lhs_scale,
                                                     rhs_scale, out);
}
